// RealImagBitNetLinearPhaseQuant_17978733101537
// MI455X (gfx1250) — matrix operand order repaired and checked by reading, not compiled, not run on hardware
//
#include <hip/hip_runtime.h>
#include <hip/hip_bf16.h>

// ---------------------------------------------------------------------------
// RealImagBitNetLinearPhaseQuant for MI455X (gfx1250, wave32, WMMA)
//
// y_re = s_re * (nx_re @ q_re^T - nx_im @ q_im^T)
// y_im = s_im * (nx_re @ q_im^T + nx_im @ q_re^T)
// with nx = LayerNorm(concat(x_re, x_im)), q in {-1,0,+1} (exact in f16).
// Matrix path: v_wmma_f32_16x16x32_f16 (f32 accumulate).
// Data path:   double-buffered LDS fed by global_load_async_to_lds_b128
//              (ASYNCcnt), overlapping DMA with the WMMA burst.
// ---------------------------------------------------------------------------

typedef __attribute__((ext_vector_type(16))) _Float16       v16h;
typedef __attribute__((ext_vector_type(16))) unsigned short v16u;
typedef __attribute__((ext_vector_type(8)))  float          v8f;

#define IN_F   4096
#define OUT_F  4096
#define NTOK   8192          // B*S = 4*2048
#define TWO_F  8192          // 2*IN_F
#define W_ELEMS (4096u*4096u)

// GEMM tiling
#define BM    64             // block tile M
#define BN    128            // block tile N
#define KC    32             // K chunk (one wmma K-depth)
#define LDSTR 40             // LDS row stride in halves (32 + 8 pad, keeps 16B align)

// workspace layout (bytes)
#define PARTIAL_OFF 0                      // 4096 blocks * 2 floats = 32 KB
#define SCALE_OFF   32768                  // 2 floats (s_re, s_im)
#define QRE_OFF     65536                  // 4096*4096 f16 = 32 MB
#define QIM_OFF     (QRE_OFF + 33554432)
#define NX_OFF      (QIM_OFF + 33554432)   // 8192*8192 f16 = 128 MB
#define YIM_ELEMS   ((size_t)NTOK * OUT_F) // 33554432 floats

union Frag {
    v16h h;
    v16u u16;
    uint4 q[2];
};

__device__ __forceinline__ v16h neg16(v16h a) {
    Frag f; f.h = a;
    f.u16 ^= (unsigned short)0x8000;       // flip sign bits of all 16 halves
    return f.h;
}

__device__ __forceinline__ v8f wmma_f16(v16h a, v16h b, v8f c) {
    // (neg_a, A, neg_b, B, c_mod, C, reuse_a, reuse_b)
    return __builtin_amdgcn_wmma_f32_16x16x32_f16(false, a, false, b, (short)0, c,
                                                  false, false);
}

// Async DMA: 16B global -> LDS, tracked by ASYNCcnt (CDNA5 §10 async ops).
// GVS mode: mem = SGPR_u64(base) + VGPR_i32(byte offset); vdst = LDS byte offset.
__device__ __forceinline__ void async_b128(unsigned lds_off, unsigned gbyte_off,
                                           const void* sbase) {
    asm volatile("global_load_async_to_lds_b128 %0, %1, %2"
                 :
                 : "v"(lds_off), "v"(gbyte_off), "s"(sbase)
                 : "memory");
}

__device__ __forceinline__ void wait_async0() {
    asm volatile("s_wait_asynccnt 0x0" ::: "memory");
}

__device__ __forceinline__ unsigned lds_off32(const void* p) {
    // flat shared pointer: high 32 bits = LDS aperture, low 32 = in-group offset
    return (unsigned)(size_t)p;
}

// ---------------------------------------------------------------------------
// Kernel 1: phase-quantize weights -> f16 {-1,0,+1}; per-block |w| partials.
// ---------------------------------------------------------------------------
__global__ void __launch_bounds__(256, 4)
quant_kernel(const float* __restrict__ wre, const float* __restrict__ wim,
             _Float16* __restrict__ qre, _Float16* __restrict__ qim,
             float* __restrict__ partial) {
    const size_t nchunks = (size_t)W_ELEMS / 4;
    float sre = 0.f, sim = 0.f;
    for (size_t c = (size_t)blockIdx.x * blockDim.x + threadIdx.x; c < nchunks;
         c += (size_t)gridDim.x * blockDim.x) {
        float4 r = ((const float4*)wre)[c];
        float4 m = ((const float4*)wim)[c];
        float rr[4] = {r.x, r.y, r.z, r.w};
        float mm[4] = {m.x, m.y, m.z, m.w};
        union { _Float16 h[4]; uint2 u; } ur, ui;
#pragma unroll
        for (int k = 0; k < 4; ++k) {
            float re = rr[k], im = mm[k];
            float are = fabsf(re), aim = fabsf(im);
            sre += are; sim += aim;
            float sgr = (re > 0.f) ? 1.f : ((re < 0.f) ? -1.f : 0.f);
            float sgi = (im > 0.f) ? 1.f : ((im < 0.f) ? -1.f : 0.f);
            bool dom = (are >= aim);
            ur.h[k] = (_Float16)(dom ? sgr : 0.f);
            ui.h[k] = (_Float16)(dom ? 0.f : sgi);
        }
        ((uint2*)qre)[c] = ur.u;
        ((uint2*)qim)[c] = ui.u;
    }
    __shared__ float red[2][256];
    int tid = threadIdx.x;
    red[0][tid] = sre; red[1][tid] = sim;
    __syncthreads();
    for (int off = 128; off > 0; off >>= 1) {
        if (tid < off) {
            red[0][tid] += red[0][tid + off];
            red[1][tid] += red[1][tid + off];
        }
        __syncthreads();
    }
    if (tid == 0) {
        partial[2 * blockIdx.x]     = red[0][0];
        partial[2 * blockIdx.x + 1] = red[1][0];
    }
}

// ---------------------------------------------------------------------------
// Kernel 2: fold 4096 partial pairs -> s_re, s_im (deterministic order).
// ---------------------------------------------------------------------------
__global__ void __launch_bounds__(256, 1)
reduce_kernel(const float* __restrict__ partial, float* __restrict__ scal) {
    int tid = threadIdx.x;
    float a = 0.f, b = 0.f;
    for (int i = tid; i < 4096; i += 256) {
        a += partial[2 * i];
        b += partial[2 * i + 1];
    }
    __shared__ float red[2][256];
    red[0][tid] = a; red[1][tid] = b;
    __syncthreads();
    for (int off = 128; off > 0; off >>= 1) {
        if (tid < off) {
            red[0][tid] += red[0][tid + off];
            red[1][tid] += red[1][tid + off];
        }
        __syncthreads();
    }
    if (tid == 0) {
        const float inv = 1.f / (float)W_ELEMS;
        scal[0] = red[0][0] * inv;   // s_re = mean|w_re|
        scal[1] = red[1][0] * inv;   // s_im = mean|w_im|
    }
}

// ---------------------------------------------------------------------------
// Kernel 3: LayerNorm over the 8192-wide concat, write f16 nx[token][8192].
// ---------------------------------------------------------------------------
__global__ void __launch_bounds__(256, 4)
ln_kernel(const float* __restrict__ xr, const float* __restrict__ xi,
          const float* __restrict__ gamma, const float* __restrict__ beta,
          _Float16* __restrict__ nx) {
    const int token = blockIdx.x;
    const int tid = threadIdx.x;
    const float* pr = xr + (size_t)token * IN_F;
    const float* pi = xi + (size_t)token * IN_F;

    float vr[16], vi[16];
    float s = 0.f, sq = 0.f;
#pragma unroll
    for (int j = 0; j < 16; ++j) {
        int f = tid + j * 256;
        float a = pr[f], b = pi[f];
        vr[j] = a; vi[j] = b;
        s  += a + b;
        sq += a * a + b * b;
    }
    __shared__ float red[2][256];
    red[0][tid] = s; red[1][tid] = sq;
    __syncthreads();
    for (int off = 128; off > 0; off >>= 1) {
        if (tid < off) {
            red[0][tid] += red[0][tid + off];
            red[1][tid] += red[1][tid + off];
        }
        __syncthreads();
    }
    const float invN = 1.f / (float)TWO_F;
    float mean = red[0][0] * invN;
    float var  = red[1][0] * invN - mean * mean;
    float rstd = rsqrtf(var + 1e-6f);

    _Float16* o = nx + (size_t)token * TWO_F;
#pragma unroll
    for (int j = 0; j < 16; ++j) {
        int f = tid + j * 256;
        o[f]        = (_Float16)((vr[j] - mean) * rstd * gamma[f] + beta[f]);
        o[IN_F + f] = (_Float16)((vi[j] - mean) * rstd * gamma[IN_F + f] + beta[IN_F + f]);
    }
}

// ---------------------------------------------------------------------------
// Kernel 4: fused complex GEMM via v_wmma_f32_16x16x32_f16.
// Block tile 64x128, 8 waves (2x4), each wave 32x32 = 2x2 16x16 tiles.
// Double-buffered LDS stages filled by global_load_async_to_lds_b128:
//   iter i:  s_wait_asynccnt 0 (stage-i fills done) -> barrier ->
//            issue stage-(i+1) DMA -> 16 WMMA from stage i.
// ---------------------------------------------------------------------------
__global__ void __launch_bounds__(256, 1)
cgemm_kernel(const _Float16* __restrict__ nx,
             const _Float16* __restrict__ qre, const _Float16* __restrict__ qim,
             const float* __restrict__ scal, float* __restrict__ out) {
    __shared__ __align__(16) _Float16 lsA[2][2][BM * LDSTR]; // [stage][re/im]
    __shared__ __align__(16) _Float16 lsB[2][2][BN * LDSTR]; // [stage][re/im]

    const int tid  = threadIdx.x;
    const int wave = tid >> 5;
    const int lane = tid & 31;
    const int waveM = wave >> 2;        // 0..1  -> 32-row strip
    const int waveN = wave & 3;         // 0..3  -> 32-col strip
    const int bm = blockIdx.y * BM;
    const int bn = blockIdx.x * BN;

    v8f acc_re[2][2], acc_im[2][2];
    const v8f vzero = {0.f, 0.f, 0.f, 0.f, 0.f, 0.f, 0.f, 0.f};
#pragma unroll
    for (int sm = 0; sm < 2; ++sm)
#pragma unroll
        for (int sn = 0; sn < 2; ++sn) { acc_re[sm][sn] = vzero; acc_im[sm][sn] = vzero; }

    // global->LDS fill assignment (per-thread 16B chunks)
    const int a_row = tid >> 2;         // 0..63
    const int a_ks  = (tid & 3) << 3;   // 0,8,16,24 halves
    const int b_row = tid >> 1;         // 0..127
    const int b_ks  = (tid & 1) << 4;   // 0,16 halves

    // LDS byte offsets of this thread's fill targets, per stage
    unsigned laRe[2], laIm[2], lbRe[2], lbIm[2];
#pragma unroll
    for (int st = 0; st < 2; ++st) {
        laRe[st] = lds_off32(&lsA[st][0][a_row * LDSTR + a_ks]);
        laIm[st] = lds_off32(&lsA[st][1][a_row * LDSTR + a_ks]);
        lbRe[st] = lds_off32(&lsB[st][0][b_row * LDSTR + b_ks]);
        lbIm[st] = lds_off32(&lsB[st][1][b_row * LDSTR + b_ks]);
    }
    const unsigned gaBase = (unsigned)(((bm + a_row) * TWO_F + a_ks) * 2); // + kc*2
    const unsigned gbBase = (unsigned)(((bn + b_row) * IN_F + b_ks) * 2);  // + kc*2

    auto issue_stage = [&](int st, int kc) {
        const unsigned ga = gaBase + (unsigned)(kc * 2);
        const unsigned gb = gbBase + (unsigned)(kc * 2);
        async_b128(laRe[st], ga,               nx);   // A_re
        async_b128(laIm[st], ga + IN_F * 2,    nx);   // A_im
        async_b128(lbRe[st], gb,               qre);  // B_re lo
        async_b128(lbRe[st] + 16, gb + 16,     qre);  // B_re hi
        async_b128(lbIm[st], gb,               qim);  // B_im lo
        async_b128(lbIm[st] + 16, gb + 16,     qim);  // B_im hi
    };

    // fragment fetch indices (16-bit A 16x32 / B 32x16 VGPR striping)
    const int am = lane & 15;
    const int ak = (lane >> 4) << 3;    // lanes 0-15: K 0..7/16..23; 16-31: K 8..15/24..31
    const int bc = lane & 15;
    const int bk = (lane >> 4) << 3;    // lanes 0-15: K 0..7/16..23; 16-31: K 8..15/24..31

    issue_stage(0, 0);                  // prologue DMA
    int cur = 0;

    for (int kc = 0; kc < IN_F; kc += KC) {
        wait_async0();                  // own cur-stage DMA complete
        __syncthreads();                // all waves: cur data visible, prev reads done
        if (kc + KC < IN_F)
            issue_stage(cur ^ 1, kc + KC);   // overlaps with WMMA below

        // ---- A fragments for this wave's two 16-row subtiles ----
        Frag fARe[2], fAIm[2];
#pragma unroll
        for (int sm = 0; sm < 2; ++sm) {
            const int r = waveM * 32 + sm * 16 + am;
            fARe[sm].q[0] = *(const uint4*)&lsA[cur][0][r * LDSTR + ak];
            fARe[sm].q[1] = *(const uint4*)&lsA[cur][0][r * LDSTR + ak + 16];
            fAIm[sm].q[0] = *(const uint4*)&lsA[cur][1][r * LDSTR + ak];
            fAIm[sm].q[1] = *(const uint4*)&lsA[cur][1][r * LDSTR + ak + 16];
        }
        // FP16 WMMA has no A/B NEG per ISA -> negate A_im via sign-bit XOR
        v16h aImNeg[2] = { neg16(fAIm[0].h), neg16(fAIm[1].h) };

#pragma unroll
        for (int sn = 0; sn < 2; ++sn) {
            const int r = waveN * 32 + sn * 16 + bc;
            Frag fBRe, fBIm;
            fBRe.q[0] = *(const uint4*)&lsB[cur][0][r * LDSTR + bk];
            fBRe.q[1] = *(const uint4*)&lsB[cur][0][r * LDSTR + bk + 16];
            fBIm.q[0] = *(const uint4*)&lsB[cur][1][r * LDSTR + bk];
            fBIm.q[1] = *(const uint4*)&lsB[cur][1][r * LDSTR + bk + 16];
#pragma unroll
            for (int sm = 0; sm < 2; ++sm) {
                acc_re[sm][sn] = wmma_f16(fARe[sm].h, fBRe.h, acc_re[sm][sn]);
                acc_re[sm][sn] = wmma_f16(aImNeg[sm], fBIm.h, acc_re[sm][sn]);
                acc_im[sm][sn] = wmma_f16(fARe[sm].h, fBIm.h, acc_im[sm][sn]);
                acc_im[sm][sn] = wmma_f16(fAIm[sm].h, fBRe.h, acc_im[sm][sn]);
            }
        }
        cur ^= 1;
    }

    // ---- epilogue: scale and store (C/D layout: vgpr e -> row, lane -> col) ----
    const float sre = scal[0];
    const float sim = scal[1];
#pragma unroll
    for (int sm = 0; sm < 2; ++sm)
#pragma unroll
        for (int sn = 0; sn < 2; ++sn) {
            const int m0 = bm + waveM * 32 + sm * 16 + ((lane >> 4) << 3);
            const int n0 = bn + waveN * 32 + sn * 16 + (lane & 15);
            float* pr = out + (size_t)m0 * OUT_F + n0;
            float* pi = pr + YIM_ELEMS;
#pragma unroll
            for (int e = 0; e < 8; ++e) {
                pr[(size_t)e * OUT_F] = acc_re[sm][sn][e] * sre;
                pi[(size_t)e * OUT_F] = acc_im[sm][sn][e] * sim;
            }
        }
}

// ---------------------------------------------------------------------------
extern "C" void kernel_launch(void* const* d_in, const int* in_sizes, int n_in,
                              void* d_out, int out_size, void* d_ws, size_t ws_size,
                              hipStream_t stream) {
    (void)in_sizes; (void)n_in; (void)out_size; (void)ws_size;

    const float* x_real = (const float*)d_in[0];
    const float* x_imag = (const float*)d_in[1];
    const float* w_re   = (const float*)d_in[2];
    const float* w_im   = (const float*)d_in[3];
    const float* gamma  = (const float*)d_in[4];
    const float* beta   = (const float*)d_in[5];
    float* out = (float*)d_out;

    char* ws = (char*)d_ws;
    float*    partial = (float*)(ws + PARTIAL_OFF);
    float*    scal    = (float*)(ws + SCALE_OFF);
    _Float16* qre     = (_Float16*)(ws + QRE_OFF);
    _Float16* qim     = (_Float16*)(ws + QIM_OFF);
    _Float16* nx      = (_Float16*)(ws + NX_OFF);

    quant_kernel<<<4096, 256, 0, stream>>>(w_re, w_im, qre, qim, partial);
    reduce_kernel<<<1, 256, 0, stream>>>(partial, scal);
    ln_kernel<<<NTOK, 256, 0, stream>>>(x_real, x_imag, gamma, beta, nx);
    cgemm_kernel<<<dim3(OUT_F / BN, NTOK / BM), 256, 0, stream>>>(nx, qre, qim, scal, out);
}
